// Decoder_69063074119893
// MI455X (gfx1250) — matrix operand order repaired and checked by reading, not compiled, not run on hardware
//
#include <hip/hip_runtime.h>
#include <hip/hip_bf16.h>
#include <cstddef>

#define HID 256
#define NTP 8
#define EMB 256
#define NC  50000
#define NV  100
#define G3  768            // 3*HID
#define MPAD 112           // NV padded to 7*16
#define KCH 512            // K-chunk for emb GEMM
#define NKCH 98            // ceil(NC/KCH)

typedef __attribute__((ext_vector_type(16))) __bf16 v16bf;
typedef __attribute__((ext_vector_type(8)))  float  v8f;

union ABFrag {
    v16bf v;
    unsigned int u[8];
};

// round-to-nearest-even f32 -> bf16 bits
__device__ __forceinline__ unsigned short f2bf(float f) {
    union { float f; unsigned int u; } c; c.f = f;
    unsigned int u = c.u;
    u += 0x7fffu + ((u >> 16) & 1u);
    return (unsigned short)(u >> 16);
}

// pack two f32 -> one dword of two bf16 (lo = first)
__device__ __forceinline__ unsigned int pack2bf(float a, float b) {
    return (unsigned int)f2bf(a) | ((unsigned int)f2bf(b) << 16);
}

__device__ __forceinline__ float sigmoidf_(float x) {
    return 1.0f / (1.0f + __expf(-x));
}

// ---------------------------------------------------------------- K1: h0 GEMV
__global__ __launch_bounds__(256) void h0_kernel(
    const float* __restrict__ pe, const float* __restrict__ Wc,
    const float* __restrict__ bc, float* __restrict__ h0)
{
    __shared__ float pl[HID * NTP];
    const int tid = threadIdx.x;
    for (int i = tid; i < HID * NTP; i += 256) pl[i] = pe[i];
    __syncthreads();
    float acc = bc[tid];
    const float* wr = Wc + (size_t)tid * (HID * NTP);
#pragma unroll 8
    for (int j = 0; j < HID * NTP; ++j) acc += wr[j] * pl[j];
    h0[tid] = acc;
}

// ------------------------------------------- K2: emb partials = prev @ X (WMMA)
// C(112 x 256) += A(112 x K) * B(K x 256), K-chunk per workgroup.
// A[t][c] = (t>=1 && t<NV) ? H[c][t-1] : 0   (prev matrix, shifted H^T)
#define XT_STRIDE 20   // dwords per e-row of X tile (32 bf16 used, padded to 40)
__global__ __launch_bounds__(512) void emb_gemm_kernel(
    const float* __restrict__ H, const float* __restrict__ X,
    float* __restrict__ part)
{
    // A tile: [row t][kk] bf16, 32 bf16 = 16 dwords per row (contiguous frags)
    __shared__ unsigned int sA[MPAD * 16];
    // X tile stored N-major: [e][kk] bf16, padded row stride for alignment
    __shared__ unsigned int sXt[EMB * XT_STRIDE];

    const int tid  = threadIdx.x;
    const int wave = tid >> 5;
    const int lane = tid & 31;
    const int half = lane >> 4;
    const int l    = lane & 15;
    const int ch    = blockIdx.x;
    const int kbase = ch * KCH;

    const v8f zero = {0.f,0.f,0.f,0.f,0.f,0.f,0.f,0.f};
    v8f acc[7];
#pragma unroll
    for (int m = 0; m < 7; ++m) acc[m] = zero;

    for (int ks = 0; ks < KCH / 32; ++ks) {
        const int k0 = kbase + ks * 32;
        // prefetch next X k-tile (32 rows x 1KB = 32KB; 512 lanes x 64B)
        {
            const int kp = k0 + 32;
            if (kp < NC) {
                const float* nx = X + (size_t)kp * EMB + tid * 16;
                __builtin_prefetch(nx, 0, 0);
            }
        }
        // stage A tile (112 x 32) as bf16
        for (int i = tid; i < MPAD * 32; i += 512) {
            const int t = i >> 5, kk = i & 31;
            const int c = k0 + kk;
            float v = 0.f;
            if (t >= 1 && t < NV && c < NC) v = H[(size_t)c * NV + (t - 1)];
            ((unsigned short*)sA)[t * 32 + kk] = f2bf(v);
        }
        // stage X tile (32 x 256) as bf16, transposed into [e][kk]
        for (int i = tid; i < 32 * EMB; i += 512) {
            const int kk = i >> 8, e = i & 255;   // consecutive tid -> consecutive e
            const int c = k0 + kk;
            const float v = (c < NC) ? X[(size_t)c * EMB + e] : 0.f;
            ((unsigned short*)sXt)[e * (2 * XT_STRIDE) + kk] = f2bf(v);
        }
        __syncthreads();

        // B fragment for this wave's N-tile: contiguous 8 dwords per lane
        ABFrag bfr;
        {
            const int e = wave * 16 + l;
            const int base = e * XT_STRIDE + 8 * half;
#pragma unroll
            for (int v = 0; v < 8; ++v) bfr.u[v] = sXt[base + v];
        }
#pragma unroll
        for (int m = 0; m < 7; ++m) {
            ABFrag af;
            const int rb = (16 * m + l) * 16 + 8 * half;
#pragma unroll
            for (int v = 0; v < 4; ++v) af.u[v] = sA[rb + v];
#pragma unroll
            for (int v = 0; v < 4; ++v) af.u[4 + v] = sA[rb + 4 + v];
            acc[m] = __builtin_amdgcn_wmma_f32_16x16x32_bf16(
                false, af.v, false, bfr.v, (short)0, acc[m], false, false);
        }
        __syncthreads();
    }

    float* p = part + (size_t)ch * (MPAD * EMB);
#pragma unroll
    for (int m = 0; m < 7; ++m) {
#pragma unroll
        for (int v = 0; v < 8; ++v) {
            const int trow = 16 * m + v + (half ? 8 : 0);
            const int e    = wave * 16 + l;
            p[trow * EMB + e] = acc[m][v];
        }
    }
}

// ----------------------------------------------- K3: reduce partials + ReLU
__global__ __launch_bounds__(256) void reduce_emb_kernel(
    const float* __restrict__ part, float* __restrict__ emb)
{
    const int i = blockIdx.x * 256 + threadIdx.x;  // < MPAD*EMB
    float s = 0.f;
    for (int ch = 0; ch < NKCH; ++ch) s += part[(size_t)ch * (MPAD * EMB) + i];
    emb[i] = s > 0.f ? s : 0.f;
}

// -------------------------------------- K3b: gates_x = emb @ W_ih^T + b_ih
__global__ __launch_bounds__(256) void gates_kernel(
    const float* __restrict__ emb, const float* __restrict__ Wih,
    const float* __restrict__ bih, float* __restrict__ gx)
{
    const int o = blockIdx.x * 256 + threadIdx.x;
    if (o >= NV * G3) return;
    const int t = o / G3, j = o % G3;
    float acc = bih[j];
    const float* er = emb + (size_t)t * EMB;
    const float* wr = Wih + (size_t)j * EMB;
#pragma unroll 8
    for (int k = 0; k < EMB; ++k) acc += er[k] * wr[k];
    gx[o] = acc;
}

// -------------------------------------------------------- K4: GRU scan
__global__ __launch_bounds__(G3) void gru_kernel(
    const float* __restrict__ h0, const float* __restrict__ gx,
    const float* __restrict__ Whh, const float* __restrict__ bhh,
    float* __restrict__ hs)
{
    __shared__ float hl[HID];
    __shared__ float gh[G3];
    const int tid = threadIdx.x;
    if (tid < HID) hl[tid] = h0[tid];
    __syncthreads();

    const float* wr = Whh + (size_t)tid * HID;
    const float bh = bhh[tid];
    for (int t = 0; t < NV; ++t) {
        float acc = bh;
#pragma unroll 8
        for (int k = 0; k < HID; ++k) acc += wr[k] * hl[k];
        gh[tid] = acc;
        __syncthreads();
        if (tid < HID) {
            const float* g = gx + (size_t)t * G3;
            const float r = sigmoidf_(g[tid] + gh[tid]);
            const float z = sigmoidf_(g[HID + tid] + gh[HID + tid]);
            const float n = tanhf(g[2 * HID + tid] + r * gh[2 * HID + tid]);
            const float hn = (1.f - z) * n + z * hl[tid];
            hl[tid] = hn;
            hs[(size_t)t * HID + tid] = hn;
        }
        __syncthreads();
    }
}

// -------------------- K5: out = sigmoid(W_out @ hs^T + b), (NC x NV), WMMA
#define HS_STRIDE 132   // dwords per t-row of hs tile (256 bf16 used, pad to 264)
__global__ __launch_bounds__(512) void out_gemm_kernel(
    const float* __restrict__ Wo, const float* __restrict__ bo,
    const float* __restrict__ hs, float* __restrict__ out)
{
    // hs^T staged N-major: [t][k] bf16, padded row stride (16B aligned rows)
    __shared__ unsigned int sB[MPAD * HS_STRIDE];

    const int tid  = threadIdx.x;
    const int wave = tid >> 5;
    const int lane = tid & 31;
    const int half = lane >> 4;
    const int l    = lane & 15;
    const int rbase = blockIdx.x * 256 + wave * 16;

    // prefetch this workgroup's W_out block (256 rows x 1KB = 256KB)
    {
        const size_t base = (size_t)blockIdx.x * 256 * HID;
#pragma unroll
        for (int r = 0; r < 4; ++r) {
            const size_t off = base + (size_t)(r * 512 + tid) * 32;
            if (off < (size_t)NC * HID) __builtin_prefetch(Wo + off, 0, 0);
        }
    }

    // stage hs (t-major, coalesced global reads, contiguous frag rows)
    for (int i = tid; i < MPAD * HID; i += 512) {
        const int t = i >> 8, k = i & 255;
        const float v = (t < NV) ? hs[(size_t)t * HID + k] : 0.f;
        ((unsigned short*)sB)[t * (2 * HS_STRIDE) + k] = f2bf(v);
    }
    __syncthreads();

    const v8f zero = {0.f,0.f,0.f,0.f,0.f,0.f,0.f,0.f};
    v8f acc[7];
#pragma unroll
    for (int nt = 0; nt < 7; ++nt) acc[nt] = zero;

    const int  row  = rbase + l;
    const bool rok  = row < NC;
    const float* wrow = Wo + (size_t)(rok ? row : 0) * HID;
    const float4 z4 = make_float4(0.f, 0.f, 0.f, 0.f);

#pragma unroll
    for (int ks = 0; ks < HID / 32; ++ks) {
        const int k0 = ks * 32;
        // A fragment: 16 bf16 from this lane's W_out row, via 4x float4 loads
        ABFrag af;
        {
            const float4* wa = (const float4*)(wrow + k0 + 16 * half);
            const float4* wb = (const float4*)(wrow + k0 + 16 * half + 8);
            const float4 g0 = rok ? wa[0] : z4;
            const float4 g1 = rok ? wa[1] : z4;
            const float4 g2 = rok ? wb[0] : z4;
            const float4 g3 = rok ? wb[1] : z4;
            af.u[0] = pack2bf(g0.x, g0.y);
            af.u[1] = pack2bf(g0.z, g0.w);
            af.u[2] = pack2bf(g1.x, g1.y);
            af.u[3] = pack2bf(g1.z, g1.w);
            af.u[4] = pack2bf(g2.x, g2.y);
            af.u[5] = pack2bf(g2.z, g2.w);
            af.u[6] = pack2bf(g3.x, g3.y);
            af.u[7] = pack2bf(g3.z, g3.w);
        }
#pragma unroll
        for (int nt = 0; nt < 7; ++nt) {
            ABFrag bfr;
            const int t = 16 * nt + l;
            const int base = t * HS_STRIDE + (k0 >> 1) + 8 * half;
#pragma unroll
            for (int v = 0; v < 8; ++v) bfr.u[v] = sB[base + v];
            acc[nt] = __builtin_amdgcn_wmma_f32_16x16x32_bf16(
                false, af.v, false, bfr.v, (short)0, acc[nt], false, false);
        }
    }

#pragma unroll
    for (int nt = 0; nt < 7; ++nt) {
        const int t = 16 * nt + l;
        if (t >= NV) continue;
#pragma unroll
        for (int v = 0; v < 8; ++v) {
            const int c = rbase + v + (half ? 8 : 0);
            if (c < NC) {
                const float x = acc[nt][v] + bo[c];
                out[(size_t)c * NV + t] = sigmoidf_(x);
            }
        }
    }
}

extern "C" void kernel_launch(void* const* d_in, const int* in_sizes, int n_in,
                              void* d_out, int out_size, void* d_ws, size_t ws_size,
                              hipStream_t stream) {
    (void)in_sizes; (void)n_in; (void)out_size; (void)ws_size;
    const float* pe  = (const float*)d_in[0];   // phenotype_embs (8,256)
    const float* H   = (const float*)d_in[1];   // (NC, NV)
    const float* X   = (const float*)d_in[2];   // (NC, EMB)
    const float* Wc  = (const float*)d_in[3];   // (HID, HID*NTP)
    const float* bc  = (const float*)d_in[4];   // (HID,)
    const float* Wih = (const float*)d_in[5];   // (3*HID, EMB)
    const float* Whh = (const float*)d_in[6];   // (3*HID, HID)
    const float* bih = (const float*)d_in[7];   // (3*HID,)
    const float* bhh = (const float*)d_in[8];   // (3*HID,)
    const float* Wo  = (const float*)d_in[9];   // (NC, HID)
    const float* bo  = (const float*)d_in[10];  // (NC,)
    float* out = (float*)d_out;                 // (NC, NV)

    float* ws     = (float*)d_ws;
    float* w_h0   = ws;                                       // 256
    float* w_part = w_h0 + 256;                               // NKCH*MPAD*EMB
    float* w_emb  = w_part + (size_t)NKCH * MPAD * EMB;       // MPAD*EMB
    float* w_gx   = w_emb + (size_t)MPAD * EMB;               // NV*G3
    float* w_hs   = w_gx + (size_t)NV * G3;                   // NV*HID

    h0_kernel<<<1, 256, 0, stream>>>(pe, Wc, bc, w_h0);
    emb_gemm_kernel<<<NKCH, 512, 0, stream>>>(H, X, w_part);
    reduce_emb_kernel<<<(MPAD * EMB) / 256, 256, 0, stream>>>(w_part, w_emb);
    gates_kernel<<<(NV * G3 + 255) / 256, 256, 0, stream>>>(w_emb, Wih, bih, w_gx);
    gru_kernel<<<1, G3, 0, stream>>>(w_h0, w_gx, Whh, bhh, w_hs);
    out_gemm_kernel<<<(NC + 255) / 256, 512, 0, stream>>>(Wo, bo, w_hs, out);
}
